// ACEFormer_58411555226322
// MI455X (gfx1250) — matrix operand order repaired and checked by reading, not compiled, not run on hardware
//
#include <hip/hip_runtime.h>
#include <hip/hip_bf16.h>
#include <math.h>

// ---------------------------------------------------------------------------
// Types for CDNA5 WMMA (wave32): D(16x16 f32) = A(16x32 bf16) x B(32x16 bf16) + C
// ---------------------------------------------------------------------------
typedef __bf16 v16bf __attribute__((ext_vector_type(16)));
typedef __bf16 v8bf  __attribute__((ext_vector_type(8)));
typedef float  v8f   __attribute__((ext_vector_type(8)));
typedef unsigned short v8us __attribute__((ext_vector_type(8)));

#define B_        8
#define L0_       1024
#define DATA_DIM_ 8
#define EMBED_    512
#define FWD_      2048
#define HEADS_    8

__device__ __forceinline__ unsigned short f32_to_bf16_rne(float f) {
    union { float f; unsigned int u; } c; c.f = f;
    unsigned int u = c.u;
    u += 0x7FFFu + ((u >> 16) & 1u);   // round-to-nearest-even
    return (unsigned short)(u >> 16);
}

// ---------------------------------------------------------------------------
// CDNA5 async memory->LDS copy (ASYNCcnt-tracked, no VGPR round-trip).
// LDS generic addresses carry the LDS byte offset in bits [31:0] (ISA 10.2).
// ---------------------------------------------------------------------------
__device__ __forceinline__ void async_load_lds_b128(void* lds_ptr, const void* gptr) {
    unsigned lds = (unsigned)(size_t)lds_ptr;
    unsigned long long ga = (unsigned long long)(size_t)gptr;
    asm volatile("global_load_async_to_lds_b128 %0, %1, off"
                 :: "v"(lds), "v"(ga) : "memory");
}
__device__ __forceinline__ void wait_asynccnt0() {
    asm volatile("s_wait_asynccnt 0x0" ::: "memory");
}

// ---------------------------------------------------------------------------
// WMMA GEMM: C[M,N] = act(A[M,K](bf16) * W[N,K]^T(bf16) + bias) (+ residual)
// Block = 128 threads = 4 waves; wave w owns the 16x16 tile at n0 = bx*64+w*16.
// K-step = 64 (two v_wmma per iteration). A and B tiles staged to LDS with
// async b128 copies; fragments are pure contiguous ds_load_b128 reads:
//   A frag: row (lane&15), halves [kh*8..+7] and [16+kh*8..+7]   (kh = lane>>4)
//   B frag: staged transposed [n][k]; column (lane&15), halves [kh*8..+7] and [16+kh*8..+7]
// LDS row stride 72 halves = 144B = 36 banks -> conflict-free b128 across rows.
// ---------------------------------------------------------------------------
#define LDS_STRIDE 72

__global__ __launch_bounds__(128) void wmma_gemm_kernel(
    const unsigned short* __restrict__ A,   // M x K, bf16 bits, row-major
    const unsigned short* __restrict__ W,   // N x K, bf16 bits, row-major
    const float* __restrict__ bias,         // N or nullptr
    const float* __restrict__ res,          // M x N or nullptr
    float* __restrict__ C,                  // M x N f32
    int M, int N, int K, int act)           // act: 0=none, 1=relu
{
    __shared__ __attribute__((aligned(16))) unsigned short As[16][LDS_STRIDE];
    __shared__ __attribute__((aligned(16))) unsigned short Bs[4][16][LDS_STRIDE];

    const int tid  = threadIdx.x;
    const int wave = tid >> 5;
    const int lane = tid & 31;
    const int m0   = blockIdx.y * 16;
    const int n0   = blockIdx.x * 64 + wave * 16;

    // A staging: 16 rows x 8 b128-chunks = 128 chunks, one per thread
    const int ar = tid >> 3;          // 0..15
    const int ac = (tid & 7) * 8;     // half offset 0..56

    const int mrow  = lane & 15;
    const int khalf = lane >> 4;
    const int ncol  = lane & 15;

    v8f acc = {};
    for (int k0 = 0; k0 < K; k0 += 64) {
        // ---- stage A tile (16 x 64 halves): async global->LDS b128 ----
        async_load_lds_b128(&As[ar][ac],
                            &A[(size_t)(m0 + ar) * K + (k0 + ac)]);
        // ---- stage this wave's B tile transposed-by-construction:
        //      Bs[w][n][k] = W[n0+n][k0+k]; rows of W contiguous -> b128
        #pragma unroll
        for (int c = 0; c < 4; ++c) {
            int chunk = lane + c * 32;      // 0..127
            int nrow  = chunk >> 3;         // 0..15
            int cc    = (chunk & 7) * 8;    // half offset
            async_load_lds_b128(&Bs[wave][nrow][cc],
                                &W[(size_t)(n0 + nrow) * K + (k0 + cc)]);
        }
        if (k0 + 64 < K)
            __builtin_prefetch(&A[(size_t)(m0 + ar) * K + (k0 + 64 + ac)], 0, 1);

        wait_asynccnt0();       // own wave's LDS writes complete
        __syncthreads();        // all waves' LDS writes visible

        #pragma unroll
        for (int s = 0; s < 2; ++s) {
            union { v16bf v; v8bf h[2]; } af, bf;
            af.h[0] = *(const v8bf*)&As[mrow][s * 32 + khalf * 8];
            af.h[1] = *(const v8bf*)&As[mrow][s * 32 + 16 + khalf * 8];
            bf.h[0] = *(const v8bf*)&Bs[wave][ncol][s * 32 + khalf * 8];
            bf.h[1] = *(const v8bf*)&Bs[wave][ncol][s * 32 + 16 + khalf * 8];
            acc = __builtin_amdgcn_wmma_f32_16x16x32_bf16(
                      false, af.v, false, bf.v, (short)0, acc, false, false);
        }
        __syncthreads();
    }

    if (n0 >= N) return;
    const int col = n0 + ncol;
    const float bv = bias ? bias[col] : 0.0f;
    #pragma unroll
    for (int r = 0; r < 8; ++r) {
        int m = m0 + r + khalf * 8;          // C/D layout: lanes16-31 hold M=r+8
        float v = acc[r] + bv;
        if (act == 1) v = v > 0.0f ? v : 0.0f;
        if (res) v += res[(size_t)m * N + col];
        C[(size_t)m * N + col] = v;
    }
}

// ---------------------------------------------------------------------------
// Elementwise / staging kernels
// ---------------------------------------------------------------------------
__global__ void cast_bf16_kernel(const float* __restrict__ in,
                                 unsigned short* __restrict__ out, long long n) {
    long long i = (long long)blockIdx.x * blockDim.x + threadIdx.x;
    if (i < n) out[i] = f32_to_bf16_rne(in[i]);
}

__global__ void cast_pad_kernel(const float* __restrict__ in,
                                unsigned short* __restrict__ out,
                                int N, int Kreal, int Kpad) {
    long long i = (long long)blockIdx.x * blockDim.x + threadIdx.x;
    long long total = (long long)N * Kpad;
    if (i >= total) return;
    int k = (int)(i % Kpad);
    int n = (int)(i / Kpad);
    out[i] = (k < Kreal) ? f32_to_bf16_rne(in[(size_t)n * Kreal + k]) : (unsigned short)0;
}

// circular im2col: out[(b*L+l)*Kpad + ci*3+t] = bf16(x[b, (l+t-1) mod L, ci])
__global__ void im2col_circ_bf16_kernel(const float* __restrict__ x,
                                        unsigned short* __restrict__ out,
                                        int Bn, int L, int Cin, int Kpad) {
    long long i = (long long)blockIdx.x * blockDim.x + threadIdx.x;
    long long total = (long long)Bn * L * Kpad;
    if (i >= total) return;
    int k = (int)(i % Kpad);
    long long row = i / Kpad;
    int l = (int)(row % L);
    int b = (int)(row / L);
    unsigned short v = 0;
    int K = Cin * 3;
    if (k < K) {
        int ci = k / 3, t = k % 3;
        int ls = l + t - 1;
        if (ls < 0) ls += L;
        if (ls >= L) ls -= L;
        v = f32_to_bf16_rne(x[((long long)b * L + ls) * Cin + ci]);
    }
    out[i] = v;
}

__global__ void add_pos_kernel(float* __restrict__ x, int Bn, int L, int d) {
    long long i = (long long)blockIdx.x * blockDim.x + threadIdx.x;
    long long total = (long long)Bn * L * d;
    if (i >= total) return;
    int c = (int)(i % d);
    long long r = i / d;
    int l = (int)(r % L);
    int pair = c & ~1;
    float div = expf(-logf(10000.0f) * (float)pair / (float)d);
    float ang = (float)l * div;
    x[i] += (c & 1) ? cosf(ang) : sinf(ang);
}

__global__ __launch_bounds__(256) void add_layernorm_kernel(
    const float* __restrict__ x, const float* __restrict__ y,
    const float* __restrict__ g, const float* __restrict__ b,
    float* __restrict__ out, int d)
{
    __shared__ float red[256];
    long long row = blockIdx.x;
    const float* xr = x + row * d;
    const float* yr = y + row * d;
    int t = threadIdx.x;
    float s = 0.f;
    for (int c = t; c < d; c += 256) s += xr[c] + yr[c];
    red[t] = s; __syncthreads();
    for (int o = 128; o > 0; o >>= 1) { if (t < o) red[t] += red[t + o]; __syncthreads(); }
    float mean = red[0] / (float)d;
    __syncthreads();
    float s2 = 0.f;
    for (int c = t; c < d; c += 256) { float v = xr[c] + yr[c] - mean; s2 += v * v; }
    red[t] = s2; __syncthreads();
    for (int o = 128; o > 0; o >>= 1) { if (t < o) red[t] += red[t + o]; __syncthreads(); }
    float inv = rsqrtf(red[0] / (float)d + 1e-5f);
    for (int c = t; c < d; c += 256) {
        float v = xr[c] + yr[c];
        out[row * d + c] = (v - mean) * inv * g[c] + b[c];
    }
}

// ---------------------------------------------------------------------------
// Attention kernels (small FLOP fraction -> VALU + LDS reductions)
// ---------------------------------------------------------------------------
__global__ __launch_bounds__(256) void ctx_mean_kernel(
    const float* __restrict__ V, float* __restrict__ ctx, int L, int d)
{
    __shared__ float red[256];
    int b = blockIdx.y, c = blockIdx.x;
    int t = threadIdx.x;
    float s = 0.f;
    for (int l = t; l < L; l += 256) s += V[((long long)b * L + l) * d + c];
    red[t] = s; __syncthreads();
    for (int o = 128; o > 0; o >>= 1) { if (t < o) red[t] += red[t + o]; __syncthreads(); }
    float mean = red[0] / (float)L;
    for (int l = t; l < L; l += 256) ctx[((long long)b * L + l) * d + c] = mean;
}

// m[b,h,q] = max_s(q.k_sampled) - sum_s(q.k_sampled)/L  (deterministic hash sampling)
__global__ void prob_m_kernel(const float* __restrict__ Q, const float* __restrict__ Kf,
                              float* __restrict__ mout, int BH, int H, int L,
                              int dh, int d, int sk) {
    long long i = (long long)blockIdx.x * blockDim.x + threadIdx.x;
    if (i >= (long long)BH * L) return;
    int q  = (int)(i % L);
    int bh = (int)(i / L);
    int b = bh / H, h = bh % H;
    const float* qp = Q + ((long long)b * L + q) * d + h * dh;
    float mx = -1e30f, sm = 0.f;
    for (int s = 0; s < sk; ++s) {
        unsigned int hsh = 2654435761u * (unsigned)q + 40503u * (unsigned)s + 12345u;
        int kidx = (int)(hsh % (unsigned)L);
        const float* kp = Kf + ((long long)b * L + kidx) * d + h * dh;
        float dot = 0.f;
        for (int j = 0; j < dh; ++j) dot += qp[j] * kp[j];
        mx = fmaxf(mx, dot);
        sm += dot;
    }
    mout[i] = mx - sm / (float)L;
}

__global__ __launch_bounds__(256) void topk_kernel(const float* __restrict__ m,
                                                   int* __restrict__ top, int L, int u) {
    __shared__ float buf[1024];
    int bh = blockIdx.x;
    for (int i = threadIdx.x; i < L; i += 256) buf[i] = m[(long long)bh * L + i];
    __syncthreads();
    if (threadIdx.x == 0) {
        for (int t = 0; t < u; ++t) {
            float best = -1e30f; int bi = 0;
            for (int i = 0; i < L; ++i)
                if (buf[i] > best) { best = buf[i]; bi = i; }
            top[(long long)bh * u + t] = bi;
            buf[bi] = -1e30f;
        }
    }
}

__global__ __launch_bounds__(128) void prob_ctx_top_kernel(
    const float* __restrict__ Q, const float* __restrict__ Kf,
    const float* __restrict__ V, const int* __restrict__ top,
    float* __restrict__ ctx, int H, int L, int dh, int d, int u)
{
    __shared__ float sc[1024];
    __shared__ float red[128];
    int t  = threadIdx.x;
    int tt = blockIdx.x % u;
    int bh = blockIdx.x / u;
    int b = bh / H, h = bh % H;
    int ql = top[(long long)bh * u + tt];
    const float* qp = Q + ((long long)b * L + ql) * d + h * dh;
    float scale = rsqrtf((float)dh);
    float lmax = -1e30f;
    for (int k = t; k < L; k += 128) {
        const float* kp = Kf + ((long long)b * L + k) * d + h * dh;
        float dot = 0.f;
        for (int j = 0; j < dh; ++j) dot += qp[j] * kp[j];
        dot *= scale;
        sc[k] = dot;
        lmax = fmaxf(lmax, dot);
    }
    red[t] = lmax; __syncthreads();
    for (int o = 64; o > 0; o >>= 1) { if (t < o) red[t] = fmaxf(red[t], red[t + o]); __syncthreads(); }
    float gmax = red[0]; __syncthreads();
    float lsum = 0.f;
    for (int k = t; k < L; k += 128) { float e = expf(sc[k] - gmax); sc[k] = e; lsum += e; }
    red[t] = lsum; __syncthreads();
    for (int o = 64; o > 0; o >>= 1) { if (t < o) red[t] += red[t + o]; __syncthreads(); }
    float inv = 1.f / red[0];
    __syncthreads();
    for (int j = t; j < dh; j += 128) {
        float o = 0.f;
        for (int k = 0; k < L; ++k) o += sc[k] * V[((long long)b * L + k) * d + h * dh + j];
        ctx[((long long)b * L + ql) * d + h * dh + j] = o * inv;
    }
}

__global__ __launch_bounds__(128) void full_attn_kernel(
    const float* __restrict__ Q, const float* __restrict__ Kf,
    const float* __restrict__ V, float* __restrict__ ctx,
    int H, int Lq, int Lk, int dh, int d)
{
    __shared__ float sc[1024];
    __shared__ float red[128];
    int t = threadIdx.x;
    int q  = blockIdx.x % Lq;
    int bh = blockIdx.x / Lq;
    int b = bh / H, h = bh % H;
    const float* qp = Q + ((long long)b * Lq + q) * d + h * dh;
    float scale = rsqrtf((float)dh);
    float lmax = -1e30f;
    for (int k = t; k < Lk; k += 128) {
        const float* kp = Kf + ((long long)b * Lk + k) * d + h * dh;
        float dot = 0.f;
        for (int j = 0; j < dh; ++j) dot += qp[j] * kp[j];
        dot *= scale;
        sc[k] = dot;
        lmax = fmaxf(lmax, dot);
    }
    red[t] = lmax; __syncthreads();
    for (int o = 64; o > 0; o >>= 1) { if (t < o) red[t] = fmaxf(red[t], red[t + o]); __syncthreads(); }
    float gmax = red[0]; __syncthreads();
    float lsum = 0.f;
    for (int k = t; k < Lk; k += 128) { float e = expf(sc[k] - gmax); sc[k] = e; lsum += e; }
    red[t] = lsum; __syncthreads();
    for (int o = 64; o > 0; o >>= 1) { if (t < o) red[t] += red[t + o]; __syncthreads(); }
    float inv = 1.f / red[0];
    __syncthreads();
    for (int j = t; j < dh; j += 128) {
        float o = 0.f;
        for (int k = 0; k < Lk; ++k) o += sc[k] * V[((long long)b * Lk + k) * d + h * dh + j];
        ctx[((long long)b * Lq + q) * d + h * dh + j] = o * inv;
    }
}

// ---------------------------------------------------------------------------
// Distill tail: BN-scale + ELU, then maxpool(L, window 3, stride 2, -inf pad)
// ---------------------------------------------------------------------------
__global__ void bn_elu_kernel(float* __restrict__ y, const float* __restrict__ g,
                              const float* __restrict__ b, long long total, int C) {
    long long i = (long long)blockIdx.x * blockDim.x + threadIdx.x;
    if (i >= total) return;
    int c = (int)(i % C);
    float invs = rsqrtf(1.0f + 1e-5f);
    float v = y[i] * invs * g[c] + b[c];
    y[i] = v > 0.f ? v : (expf(v) - 1.f);
}

__global__ void maxpool_kernel(const float* __restrict__ y, float* __restrict__ out,
                               int Bn, int L, int C) {
    long long i = (long long)blockIdx.x * blockDim.x + threadIdx.x;
    int Lo = L / 2;
    long long total = (long long)Bn * Lo * C;
    if (i >= total) return;
    int c  = (int)(i % C);
    long long r = i / C;
    int lo = (int)(r % Lo);
    int b  = (int)(r / Lo);
    float mx = -1e30f;
    for (int t = -1; t <= 1; ++t) {
        int li = 2 * lo + t;
        if (li >= 0 && li < L)
            mx = fmaxf(mx, y[((long long)b * L + li) * C + c]);
    }
    out[i] = mx;
}

__global__ void fc_kernel(const float* __restrict__ x, const float* __restrict__ w,
                          const float* __restrict__ b, float* __restrict__ out,
                          int M, int K) {
    int m = blockIdx.x * blockDim.x + threadIdx.x;
    if (m >= M) return;
    float s = b[0];
    for (int k = 0; k < K; ++k) s += x[(long long)m * K + k] * w[k];
    out[m] = s;
}

// ---------------------------------------------------------------------------
// Host-side helpers
// ---------------------------------------------------------------------------
static inline void launch_cast(const float* a, unsigned short* o, long long n, hipStream_t s) {
    long long blocks = (n + 255) / 256;
    cast_bf16_kernel<<<dim3((unsigned)blocks), 256, 0, s>>>(a, o, n);
}
static inline void launch_gemm(const unsigned short* abf, const float* W, const float* bias,
                               const float* res, float* C, int M, int N, int K, int act,
                               unsigned short* wbf, hipStream_t s) {
    launch_cast(W, wbf, (long long)N * K, s);
    dim3 g(N / 64, M / 16);
    wmma_gemm_kernel<<<g, 128, 0, s>>>(abf, wbf, bias, res, C, M, N, K, act);
}

struct CrossP {
    const float *qw,*qb,*kw,*kb,*vw,*vb,*ow,*ob;
    const float *f1w,*f1b,*f2w,*f2b,*n1g,*n1b,*n2g,*n2b;
};

extern "C" void kernel_launch(void* const* d_in, const int* in_sizes, int n_in,
                              void* d_out, int out_size, void* d_ws, size_t ws_size,
                              hipStream_t stream) {
    (void)in_sizes; (void)n_in; (void)out_size; (void)ws_size;

    // ---- parse inputs in setup_inputs() dict-insertion order ----
    int pi = 0;
    auto F = [&]() -> const float* { return (const float*)d_in[pi++]; };
    const float* data     = F();   // (B, L, 8)
    const float* expand_w = F();   // (512, 8, 3)
    CrossP disC[3];
    const float *dw[3], *db[3], *bg[3], *bb[3];
    auto RC = [&](CrossP& c) {
        c.qw=F(); c.qb=F(); c.kw=F(); c.kb=F(); c.vw=F(); c.vb=F(); c.ow=F(); c.ob=F();
        c.f1w=F(); c.f1b=F(); c.f2w=F(); c.f2b=F();
        c.n1g=F(); c.n1b=F(); c.n2g=F(); c.n2b=F();
    };
    for (int i = 0; i < 3; ++i) { RC(disC[i]); dw[i]=F(); db[i]=F(); bg[i]=F(); bb[i]=F(); }
    CrossP attnC[2];
    for (int i = 0; i < 2; ++i) RC(attnC[i]);
    const float* fcw = F();
    const float* fcb = F();

    // ---- carve workspace ----
    const long long M0 = (long long)B_ * L0_;   // 8192
    char* wp = (char*)d_ws;
    auto carve = [&](size_t bytes) -> void* {
        void* p = wp; wp += (bytes + 255) & ~(size_t)255; return p;
    };
    float* x    = (float*)carve(sizeof(float) * M0 * EMBED_);
    float* t1   = (float*)carve(sizeof(float) * M0 * FWD_);
    float* t2   = (float*)carve(sizeof(float) * M0 * EMBED_);
    float* Qb   = (float*)carve(sizeof(float) * M0 * EMBED_);
    float* Kb   = (float*)carve(sizeof(float) * M0 * EMBED_);
    float* Vb   = (float*)carve(sizeof(float) * M0 * EMBED_);
    float* ctx  = (float*)carve(sizeof(float) * M0 * EMBED_);
    float* dout = (float*)carve(sizeof(float) * 1024 * 64);
    float* mb   = (float*)carve(sizeof(float) * B_ * HEADS_ * L0_);
    int*   topb = (int*)  carve(sizeof(int) * B_ * HEADS_ * 64);
    unsigned short* abf = (unsigned short*)carve(sizeof(unsigned short) * M0 * FWD_);
    unsigned short* wbf = (unsigned short*)carve(sizeof(unsigned short) * (size_t)FWD_ * EMBED_);

    // ---- expand conv (circular, no bias) as im2col + WMMA GEMM; + pos embedding ----
    {
        const int KP = 64;   // pad K=24 up to the GEMM's K-step
        long long tot = M0 * KP;
        im2col_circ_bf16_kernel<<<dim3((unsigned)((tot + 255) / 256)), 256, 0, stream>>>(
            data, abf, B_, L0_, DATA_DIM_, KP);
        long long wt = (long long)EMBED_ * KP;
        cast_pad_kernel<<<dim3((unsigned)((wt + 255) / 256)), 256, 0, stream>>>(
            expand_w, wbf, EMBED_, DATA_DIM_ * 3, KP);
        dim3 g(EMBED_ / 64, (unsigned)(M0 / 16));
        wmma_gemm_kernel<<<g, 128, 0, stream>>>(abf, wbf, nullptr, nullptr, x,
                                                (int)M0, EMBED_, KP, 0);
        long long pt = M0 * EMBED_;
        add_pos_kernel<<<dim3((unsigned)((pt + 255) / 256)), 256, 0, stream>>>(x, B_, L0_, EMBED_);
    }

    // ---- 3 distilling layers (ProbSparse self-attn + FFN + conv/BN/ELU/maxpool) ----
    int L = L0_, d = EMBED_;
    for (int i = 0; i < 3; ++i) {
        const int M = B_ * L, H = HEADS_, dh = d / H;
        const CrossP& c = disC[i];

        launch_cast(x, abf, (long long)M * d, stream);
        launch_gemm(abf, c.qw, c.qb, nullptr, Qb, M, d, d, 0, wbf, stream);
        launch_gemm(abf, c.kw, c.kb, nullptr, Kb, M, d, d, 0, wbf, stream);
        launch_gemm(abf, c.vw, c.vb, nullptr, Vb, M, d, d, 0, wbf, stream);

        int ck = (int)ceil(log((double)L));
        int sk = 5 * ck; if (sk > L) sk = L;
        int u  = sk;
        ctx_mean_kernel<<<dim3(d, B_), 256, 0, stream>>>(Vb, ctx, L, d);
        {
            long long tot = (long long)B_ * H * L;
            prob_m_kernel<<<dim3((unsigned)((tot + 255) / 256)), 256, 0, stream>>>(
                Qb, Kb, mb, B_ * H, H, L, dh, d, sk);
        }
        topk_kernel<<<B_ * H, 256, 0, stream>>>(mb, topb, L, u);
        prob_ctx_top_kernel<<<B_ * H * u, 128, 0, stream>>>(Qb, Kb, Vb, topb, ctx, H, L, dh, d, u);

        launch_cast(ctx, abf, (long long)M * d, stream);
        launch_gemm(abf, c.ow, c.ob, nullptr, t2, M, d, d, 0, wbf, stream);
        add_layernorm_kernel<<<M, 256, 0, stream>>>(x, t2, c.n1g, c.n1b, x, d);

        launch_cast(x, abf, (long long)M * d, stream);
        launch_gemm(abf, c.f1w, c.f1b, nullptr, t1, M, FWD_, d, 1, wbf, stream);
        launch_cast(t1, abf, (long long)M * FWD_, stream);
        launch_gemm(abf, c.f2w, c.f2b, nullptr, t2, M, d, FWD_, 0, wbf, stream);
        add_layernorm_kernel<<<M, 256, 0, stream>>>(x, t2, c.n2g, c.n2b, x, d);

        // distill: circular conv (d -> d/2) via im2col + WMMA, BN-scale+ELU, maxpool/2
        const int K3 = 3 * d, C2 = d / 2;   // K3 in {1536,768,384}: multiples of 64
        {
            long long tot = (long long)M * K3;
            im2col_circ_bf16_kernel<<<dim3((unsigned)((tot + 255) / 256)), 256, 0, stream>>>(
                x, abf, B_, L, d, K3);
        }
        launch_gemm(abf, dw[i], db[i], nullptr, t2, M, C2, K3, 0, wbf, stream);
        {
            long long tot = (long long)M * C2;
            bn_elu_kernel<<<dim3((unsigned)((tot + 255) / 256)), 256, 0, stream>>>(
                t2, bg[i], bb[i], tot, C2);
        }
        {
            long long tot = (long long)B_ * (L / 2) * C2;
            maxpool_kernel<<<dim3((unsigned)((tot + 255) / 256)), 256, 0, stream>>>(
                t2, x, B_, L, C2);
        }
        L /= 2; d /= 2;
    }

    // ---- save dis_out (cross source for the full-attention layers) ----
    const int Mf = B_ * L;                     // 1024, d == 64
    hipMemcpyAsync(dout, x, (size_t)Mf * d * sizeof(float),
                   hipMemcpyDeviceToDevice, stream);

    // ---- 2 full-attention cross layers ----
    const int FFN_F = (EMBED_ >> 2) * 2;       // 256
    for (int i = 0; i < 2; ++i) {
        const CrossP& c = attnC[i];
        const int dh = d / HEADS_;
        launch_cast(x, abf, (long long)Mf * d, stream);
        launch_gemm(abf, c.qw, c.qb, nullptr, Qb, Mf, d, d, 0, wbf, stream);
        launch_cast(dout, abf, (long long)Mf * d, stream);
        launch_gemm(abf, c.kw, c.kb, nullptr, Kb, Mf, d, d, 0, wbf, stream);
        launch_gemm(abf, c.vw, c.vb, nullptr, Vb, Mf, d, d, 0, wbf, stream);

        full_attn_kernel<<<B_ * HEADS_ * L, 128, 0, stream>>>(Qb, Kb, Vb, ctx,
                                                              HEADS_, L, L, dh, d);

        launch_cast(ctx, abf, (long long)Mf * d, stream);
        launch_gemm(abf, c.ow, c.ob, nullptr, t2, Mf, d, d, 0, wbf, stream);
        add_layernorm_kernel<<<Mf, 256, 0, stream>>>(x, t2, c.n1g, c.n1b, x, d);

        launch_cast(x, abf, (long long)Mf * d, stream);
        launch_gemm(abf, c.f1w, c.f1b, nullptr, t1, Mf, FFN_F, d, 1, wbf, stream);
        launch_cast(t1, abf, (long long)Mf * FFN_F, stream);
        launch_gemm(abf, c.f2w, c.f2b, nullptr, t2, Mf, d, FFN_F, 0, wbf, stream);
        add_layernorm_kernel<<<Mf, 256, 0, stream>>>(x, t2, c.n2g, c.n2b, x, d);
    }

    // ---- final FC (out dim 1) ----
    fc_kernel<<<(Mf + 255) / 256, 256, 0, stream>>>(x, fcw, fcb, (float*)d_out, Mf, d);
}
